// PointNetSetAbstractionMsg_66254165508131
// MI455X (gfx1250) — compile-verified, matrix operand order checked by reading, not run on hardware
//
#include <hip/hip_runtime.h>
#include <hip/hip_bf16.h>

typedef __attribute__((ext_vector_type(16))) _Float16 v16h;
typedef __attribute__((ext_vector_type(8)))  float    v8f;

#define NBATCH 4
#define NPTS   4096
#define NSEED  1024
#define INCH   64
#define FEATC  320

// ---------------------------------------------------------------------------
// Param prep: fold BN (fixed stats) + bias into per-channel scale/shift,
// convert weights to f16 padded row-major (cout, cinp).
// ---------------------------------------------------------------------------
__global__ void prep_layer(const float* __restrict__ W, const float* __restrict__ bias,
                           const float* __restrict__ beta, const float* __restrict__ gamma,
                           const float* __restrict__ mean, const float* __restrict__ var,
                           _Float16* __restrict__ Wh, float* __restrict__ aff,
                           int cout, int cin, int cinp)
{
    int i = blockIdx.x * blockDim.x + threadIdx.x;
    int tot = cout * cinp;
    if (i < tot) {
        int r = i / cinp, c = i % cinp;
        Wh[i] = (c < cin) ? (_Float16)W[r * cin + c] : (_Float16)0.f;
    }
    if (i < cout) {
        float sc = gamma[i] * rsqrtf(var[i] + 1e-5f);
        aff[i] = sc;
        aff[cout + i] = sc * (bias[i] - mean[i]) + beta[i];
    }
}

// ---------------------------------------------------------------------------
// Farthest point sampling: one block per batch. xyz cached in LDS, running
// min-dist in registers (16 pts/thread), block argmax with low-index tiebreak.
// Emits carry-before-update like the reference scan (idx[0] == 0).
// ---------------------------------------------------------------------------
__global__ __launch_bounds__(256)
void fps_kernel(const float* __restrict__ xyz, float* __restrict__ newxyz,
                int* __restrict__ fidx)
{
    int b = blockIdx.x, t = threadIdx.x;
    __shared__ float sx[NPTS], sy[NPTS], sz[NPTS];
    __shared__ float rval[256];
    __shared__ int   ridx[256];
    __shared__ int   sfar;
    const float* base = xyz + (long)b * NPTS * 3;
    for (int p = t; p < NPTS; p += 256) {
        sx[p] = base[p * 3 + 0];
        sy[p] = base[p * 3 + 1];
        sz[p] = base[p * 3 + 2];
    }
    float dist[16];
#pragma unroll
    for (int j = 0; j < 16; ++j) dist[j] = 1e10f;
    if (t == 0) sfar = 0;
    __syncthreads();

    for (int it = 0; it < NSEED; ++it) {
        int far = sfar;
        if (t == 0) {
            fidx[b * NSEED + it] = far;
            long o = ((long)b * NSEED + it) * 3;
            newxyz[o + 0] = sx[far];
            newxyz[o + 1] = sy[far];
            newxyz[o + 2] = sz[far];
        }
        float cx = sx[far], cy = sy[far], cz = sz[far];
        float best = -1.f; int bi = t * 16;
#pragma unroll
        for (int j = 0; j < 16; ++j) {
            int p = t * 16 + j;
            float dx = sx[p] - cx, dy = sy[p] - cy, dz = sz[p] - cz;
            float d = dx * dx + dy * dy + dz * dz;
            dist[j] = fminf(dist[j], d);
            if (dist[j] > best) { best = dist[j]; bi = p; }
        }
        rval[t] = best; ridx[t] = bi;
        __syncthreads();
        for (int off = 128; off > 0; off >>= 1) {
            if (t < off) {
                float v2 = rval[t + off]; int i2 = ridx[t + off];
                if (v2 > rval[t] || (v2 == rval[t] && i2 < ridx[t])) {
                    rval[t] = v2; ridx[t] = i2;
                }
            }
            __syncthreads();
        }
        if (t == 0) sfar = ridx[0];
        __syncthreads();
    }
}

// ---------------------------------------------------------------------------
// Ball query: one wave32 per (b,s). Ordered append via ballot preserves index
// order (== reference sort-then-truncate); pad with first hit.
// ---------------------------------------------------------------------------
__global__ __launch_bounds__(256)
void ball_query(const float* __restrict__ xyz, const float* __restrict__ newxyz,
                int* __restrict__ out_idx, float r2, int K)
{
    int w = blockIdx.x * 8 + (threadIdx.x >> 5);
    int lane = threadIdx.x & 31;
    int b = w >> 10;
    float cx = newxyz[(long)w * 3 + 0];
    float cy = newxyz[(long)w * 3 + 1];
    float cz = newxyz[(long)w * 3 + 2];
    const float* base = xyz + (long)b * NPTS * 3;
    int count = 0, first = 0;
    bool have_first = false;
    for (int ch = 0; ch < NPTS / 32; ++ch) {
        int p = ch * 32 + lane;
        float dx = base[p * 3 + 0] - cx;
        float dy = base[p * 3 + 1] - cy;
        float dz = base[p * 3 + 2] - cz;
        bool pass = (dx * dx + dy * dy + dz * dz) <= r2;
        unsigned mask = __builtin_amdgcn_ballot_w32(pass);
        if (mask) {
            if (!have_first) { first = ch * 32 + __builtin_ctz(mask); have_first = true; }
            if (pass) {
                int pos = count + __popc(mask & ((1u << lane) - 1u));
                if (pos < K) out_idx[(long)w * K + pos] = p;
            }
            count += __popc(mask);
            if (count >= K) break;
        }
    }
    int start = count < K ? count : K;
    for (int pos = start + lane; pos < K; pos += 32)
        out_idx[(long)w * K + pos] = first;
}

// ---------------------------------------------------------------------------
// Gather grouped features (relative xyz ++ point feats) -> f16, K-padded to 96.
// One block per row (b,s,j); threads cover the 96 channels.
// ---------------------------------------------------------------------------
__global__ __launch_bounds__(128)
void group_feats(const float* __restrict__ xyz, const float* __restrict__ points,
                 const float* __restrict__ newxyz, const int* __restrict__ idx,
                 _Float16* __restrict__ X, int K)
{
    long row = blockIdx.x;
    int c = threadIdx.x;
    if (c >= 96) return;
    int j = (int)(row % K);
    long bs = row / K;
    int b = (int)(bs >> 10);
    int pi = idx[bs * K + j];
    float v;
    if (c < 3)
        v = xyz[((long)b * NPTS + pi) * 3 + c] - newxyz[bs * 3 + c];
    else if (c < 3 + INCH)
        v = points[((long)b * NPTS + pi) * INCH + (c - 3)];
    else
        v = 0.f;
    X[row * 96 + c] = (_Float16)v;
}

// ---------------------------------------------------------------------------
// Tall-skinny GEMM + fused BN/bias/ReLU via v_wmma_f32_16x16x32_f16.
// Y[rows,COUT] = relu(X[rows,CIN] * Wh[COUT,CIN]^T * scale + shift)
// Block = 256 (8 waves), each wave owns a 16-row tile; A frags hoisted
// across the N loop. EXEC is all-ones throughout (no divergence).
// ---------------------------------------------------------------------------
template <int KT, int NT>
__global__ __launch_bounds__(256)
void gemm_bn_relu(const _Float16* __restrict__ X, const _Float16* __restrict__ Wh,
                  const float* __restrict__ aff, _Float16* __restrict__ Y)
{
    constexpr int CIN = KT * 32, COUT = NT * 16;
    const int lane = threadIdx.x & 31;
    const int wave = threadIdx.x >> 5;
    const long row0 = (long)blockIdx.x * 128 + wave * 16;
    const int m = lane & 15;
    const int hi = lane >> 4;

    // A fragments: 16-bit A 16x32 layout -> per lane, 8 contiguous halves at
    // K = kt*32 + hi*8 (VGPR0-3) and K = kt*32 + 16 + hi*8 (VGPR4-7).
    v16h a[KT];
    {
        const _Float16* xrow = X + (row0 + m) * CIN;
#pragma unroll
        for (int kt = 0; kt < KT; ++kt) {
            const _Float16* p = xrow + kt * 32 + hi * 8;
            union { uint4 u[2]; v16h h; } cvt;
            cvt.u[0] = *(const uint4*)(p);
            cvt.u[1] = *(const uint4*)(p + 16);
            a[kt] = cvt.h;
        }
    }

#pragma unroll
    for (int nt = 0; nt < NT; ++nt) {
        v8f c = {};
        const _Float16* wrow = Wh + (nt * 16 + m) * CIN;
#pragma unroll
        for (int kt = 0; kt < KT; ++kt) {
            const _Float16* p = wrow + kt * 32 + hi * 8;
            union { uint4 u[2]; v16h h; } cvt;
            cvt.u[0] = *(const uint4*)(p);
            cvt.u[1] = *(const uint4*)(p + 16);
            c = __builtin_amdgcn_wmma_f32_16x16x32_f16(
                    false, a[kt], false, cvt.h, (short)0, c, false, false);
        }
        // C/D layout: col = lane&15, row = v + hi*8
        const int col = nt * 16 + m;
        const float s = aff[col];
        const float t = aff[COUT + col];
#pragma unroll
        for (int v = 0; v < 8; ++v) {
            long r = row0 + v + hi * 8;
            float y = c[v] * s + t;
            y = y > 0.f ? y : 0.f;
            Y[r * COUT + col] = (_Float16)y;
        }
    }
}

// ---------------------------------------------------------------------------
// Max over the K grouped neighbors -> feat[b, cbase+c, s] (f32, NCW layout).
// ---------------------------------------------------------------------------
__global__ void group_maxpool(const _Float16* __restrict__ Y, float* __restrict__ feat,
                              int K, int COUT, int cbase)
{
    long total = (long)NBATCH * NSEED * COUT;
    for (long i = (long)blockIdx.x * blockDim.x + threadIdx.x; i < total;
         i += (long)gridDim.x * blockDim.x) {
        int c = (int)(i % COUT);
        long bs = i / COUT;
        const _Float16* p = Y + bs * K * COUT + c;
        float mx = (float)p[0];
        for (int j = 1; j < K; ++j) mx = fmaxf(mx, (float)p[(long)j * COUT]);
        int b = (int)(bs >> 10), s = (int)(bs & 1023);
        feat[((long)b * FEATC + cbase + c) * NSEED + s] = mx;
    }
}

// ---------------------------------------------------------------------------
// CBAM: channel pooling, attention (length-1 conv => center-tap matvecs),
// apply; then spatial stats + 3x3 conv + apply.
// ---------------------------------------------------------------------------
__global__ void chan_pool(const float* __restrict__ feat, float* __restrict__ avg,
                          float* __restrict__ mx)
{
    int i = blockIdx.x * blockDim.x + threadIdx.x;
    if (i >= NBATCH * FEATC) return;
    const float* p = feat + (long)i * NSEED;
    float s = 0.f, m = -1e30f;
    for (int t = 0; t < NSEED; ++t) { float v = p[t]; s += v; m = fmaxf(m, v); }
    avg[i] = s * (1.f / NSEED);
    mx[i] = m;
}

struct CAW { const float* W1[6]; const float* W2[6]; };  // [pool*3 + branch]

__global__ __launch_bounds__(64)
void channel_att(CAW w, const float* __restrict__ avg, const float* __restrict__ mx,
                 float* __restrict__ ca)
{
    int b = blockIdx.x, t = threadIdx.x;
    __shared__ float h[20];
    __shared__ float acc[FEATC];
    __shared__ float pool[FEATC];
    for (int c = t; c < FEATC; c += 64) acc[c] = 0.f;
    for (int p = 0; p < 2; ++p) {
        const float* src = (p == 0 ? avg : mx) + b * FEATC;
        for (int c = t; c < FEATC; c += 64) pool[c] = src[c];
        __syncthreads();
        for (int br = 0; br < 3; ++br) {
            int k = 2 * br + 1, ctr = br;
            const float* W1 = w.W1[p * 3 + br];
            const float* W2 = w.W2[p * 3 + br];
            if (t < 20) {
                float s = 0.f;
                for (int c = 0; c < FEATC; ++c) s += W1[(t * FEATC + c) * k + ctr] * pool[c];
                h[t] = s > 0.f ? s : 0.f;
            }
            __syncthreads();
            for (int c = t; c < FEATC; c += 64) {
                float s = 0.f;
                for (int q = 0; q < 20; ++q) s += W2[(c * 20 + q) * k + ctr] * h[q];
                acc[c] += s;
            }
            __syncthreads();
        }
    }
    for (int c = t; c < FEATC; c += 64)
        ca[b * FEATC + c] = 1.f / (1.f + __expf(-acc[c]));
}

__global__ void apply_ca(float* __restrict__ feat, const float* __restrict__ ca)
{
    long total = (long)NBATCH * FEATC * NSEED;
    for (long i = (long)blockIdx.x * blockDim.x + threadIdx.x; i < total;
         i += (long)gridDim.x * blockDim.x)
        feat[i] *= ca[i / NSEED];
}

__global__ void spatial_stats(const float* __restrict__ feat, float* __restrict__ spa)
{
    int i = blockIdx.x * blockDim.x + threadIdx.x;
    if (i >= NBATCH * NSEED) return;
    int b = i >> 10, s = i & 1023;
    const float* p = feat + (long)b * FEATC * NSEED + s;
    float mxv = -1e30f, mnv = 1e30f, sum = 0.f;
    for (int c = 0; c < FEATC; ++c) {
        float v = p[(long)c * NSEED];
        mxv = fmaxf(mxv, v); mnv = fminf(mnv, v); sum += v;
    }
    float* o = spa + (long)b * 3 * NSEED;
    o[s] = mxv;
    o[NSEED + s] = sum * (1.f / FEATC);
    o[2 * NSEED + s] = mnv;
}

__global__ void spatial_apply(const float* __restrict__ feat, const float* __restrict__ spa,
                              const float* __restrict__ Wsp, float* __restrict__ out)
{
    int i = blockIdx.x * blockDim.x + threadIdx.x;
    if (i >= NBATCH * NSEED) return;
    int b = i >> 10, s = i & 1023;
    const float* sp = spa + (long)b * 3 * NSEED;
    float acc = 0.f;
    for (int ch = 0; ch < 3; ++ch)
        for (int t = 0; t < 3; ++t) {
            int ss = s + t - 1;
            if (ss >= 0 && ss < NSEED) acc += Wsp[ch * 3 + t] * sp[ch * NSEED + ss];
        }
    float sa = 1.f / (1.f + __expf(-acc));
    const float* f = feat + (long)b * FEATC * NSEED + s;
    float* o = out + (long)b * FEATC * NSEED + s;
    for (int c = 0; c < FEATC; ++c) o[(long)c * NSEED] = f[(long)c * NSEED] * sa;
}

// ---------------------------------------------------------------------------
// Host
// ---------------------------------------------------------------------------
struct LayerDesc { int din; int cout, cin, cinp; };

static void launch_gemm(int cinp, int cout, long rows, const _Float16* X,
                        const _Float16* Wh, const float* aff, _Float16* Y,
                        hipStream_t st)
{
    dim3 g((unsigned)(rows / 128)), blk(256);
    int kt = cinp / 32, nt = cout / 16;
#define GEMM_CASE(KT, NT) \
    if (kt == KT && nt == NT) { gemm_bn_relu<KT, NT><<<g, blk, 0, st>>>(X, Wh, aff, Y); return; }
    GEMM_CASE(3, 2) GEMM_CASE(1, 2) GEMM_CASE(1, 4) GEMM_CASE(3, 4)
    GEMM_CASE(2, 4) GEMM_CASE(2, 8) GEMM_CASE(2, 6) GEMM_CASE(3, 8)
#undef GEMM_CASE
}

extern "C" void kernel_launch(void* const* d_in, const int* in_sizes, int n_in,
                              void* d_out, int out_size, void* d_ws, size_t ws_size,
                              hipStream_t stream)
{
    (void)in_sizes; (void)n_in; (void)out_size; (void)ws_size;
    // Input order: xyz, points, then params pytree (sorted keys):
    // cbam{avg[0..2]{W1,W2}, max[0..2]{W1,W2}, spatial}, scales[0..2][layer]{W,b,beta,gamma,mean,var}
    const float* xyz = (const float*)d_in[0];
    const float* points = (const float*)d_in[1];
    const float* spatialW = (const float*)d_in[14];

    static const LayerDesc LD[9] = {
        {15, 32, 67, 96}, {21, 32, 32, 32}, {27, 64, 32, 32},
        {33, 64, 67, 96}, {39, 64, 64, 64}, {45, 128, 64, 64},
        {51, 64, 67, 96}, {57, 96, 64, 64}, {63, 128, 96, 96}};

    // Workspace carve-out
    char* ws = (char*)d_ws;
    size_t cur = 0;
    auto alloc = [&](size_t bytes) -> char* {
        char* p = ws + cur;
        cur = (cur + bytes + 255) & ~(size_t)255;
        return p;
    };
    int* fps_idx = (int*)alloc(NBATCH * NSEED * 4);
    int* ballidx[3];
    const int Ks[3] = {16, 32, 128};
    const float r2s[3] = {0.01f, 0.04f, 0.16f};
    const int cbase[3] = {0, 64, 192};
    for (int sc = 0; sc < 3; ++sc)
        ballidx[sc] = (int*)alloc((size_t)NBATCH * NSEED * Ks[sc] * 4);
    float* feat = (float*)alloc((size_t)NBATCH * FEATC * NSEED * 4);
    float* avgp = (float*)alloc(NBATCH * FEATC * 4);
    float* maxp = (float*)alloc(NBATCH * FEATC * 4);
    float* cab  = (float*)alloc(NBATCH * FEATC * 4);
    float* spa  = (float*)alloc(NBATCH * 3 * NSEED * 4);
    _Float16* wh[9]; float* aff[9];
    for (int l = 0; l < 9; ++l) {
        wh[l] = (_Float16*)alloc((size_t)LD[l].cout * LD[l].cinp * 2);
        aff[l] = (float*)alloc((size_t)2 * LD[l].cout * 4);
    }
    const long rowsMax = (long)NBATCH * NSEED * 128;
    _Float16* bufA = (_Float16*)alloc((size_t)rowsMax * 96 * 2);
    _Float16* bufB = (_Float16*)alloc((size_t)rowsMax * 128 * 2);

    // Prep weights (W->f16 padded, BN folded to scale/shift)
    for (int l = 0; l < 9; ++l) {
        const LayerDesc& L = LD[l];
        int tot = L.cout * L.cinp;
        prep_layer<<<(tot + 255) / 256, 256, 0, stream>>>(
            (const float*)d_in[L.din + 0], (const float*)d_in[L.din + 1],
            (const float*)d_in[L.din + 2], (const float*)d_in[L.din + 3],
            (const float*)d_in[L.din + 4], (const float*)d_in[L.din + 5],
            wh[l], aff[l], L.cout, L.cin, L.cinp);
    }

    // FPS -> new_xyz (first 12288 floats of d_out)
    float* newxyz = (float*)d_out;
    float* outFinal = (float*)d_out + (size_t)NBATCH * NSEED * 3;
    fps_kernel<<<NBATCH, 256, 0, stream>>>(xyz, newxyz, fps_idx);

    // Per-scale: ball query -> gather -> 3x (WMMA GEMM + BN + ReLU) -> max-pool
    for (int sc = 0; sc < 3; ++sc) {
        long rows = (long)NBATCH * NSEED * Ks[sc];
        ball_query<<<(NBATCH * NSEED) / 8, 256, 0, stream>>>(
            xyz, newxyz, ballidx[sc], r2s[sc], Ks[sc]);
        group_feats<<<(unsigned)rows, 128, 0, stream>>>(
            xyz, points, newxyz, ballidx[sc], bufA, Ks[sc]);
        const LayerDesc& L0 = LD[sc * 3 + 0];
        const LayerDesc& L1 = LD[sc * 3 + 1];
        const LayerDesc& L2 = LD[sc * 3 + 2];
        launch_gemm(L0.cinp, L0.cout, rows, bufA, wh[sc * 3 + 0], aff[sc * 3 + 0], bufB, stream);
        launch_gemm(L1.cinp, L1.cout, rows, bufB, wh[sc * 3 + 1], aff[sc * 3 + 1], bufA, stream);
        launch_gemm(L2.cinp, L2.cout, rows, bufA, wh[sc * 3 + 2], aff[sc * 3 + 2], bufB, stream);
        long total = (long)NBATCH * NSEED * L2.cout;
        group_maxpool<<<(unsigned)((total + 255) / 256), 256, 0, stream>>>(
            bufB, feat, Ks[sc], L2.cout, cbase[sc]);
    }

    // CBAM channel attention
    chan_pool<<<(NBATCH * FEATC + 255) / 256, 256, 0, stream>>>(feat, avgp, maxp);
    CAW cw;
    for (int br = 0; br < 3; ++br) {
        cw.W1[br]     = (const float*)d_in[2 + 2 * br];      // avg W1
        cw.W2[br]     = (const float*)d_in[3 + 2 * br];      // avg W2
        cw.W1[3 + br] = (const float*)d_in[8 + 2 * br];      // max W1
        cw.W2[3 + br] = (const float*)d_in[9 + 2 * br];      // max W2
    }
    channel_att<<<NBATCH, 64, 0, stream>>>(cw, avgp, maxp, cab);
    {
        long total = (long)NBATCH * FEATC * NSEED;
        apply_ca<<<(unsigned)((total + 255) / 256), 256, 0, stream>>>(feat, cab);
    }

    // CBAM spatial attention + final output
    spatial_stats<<<(NBATCH * NSEED + 255) / 256, 256, 0, stream>>>(feat, spa);
    spatial_apply<<<(NBATCH * NSEED + 255) / 256, 256, 0, stream>>>(feat, spa, spatialW, outFinal);
}
